// WordAttention_42949672960495
// MI455X (gfx1250) — compile-verified, matrix operand order checked by reading, not run on hardware
//
#include <hip/hip_runtime.h>
#include <math.h>

// ---------------- types for WMMA operands ----------------
typedef __bf16  v16bf __attribute__((ext_vector_type(16)));
typedef float   v8f   __attribute__((ext_vector_type(8)));
typedef unsigned int u32x4 __attribute__((ext_vector_type(4)));

union BF16x16 {
    v16bf bf;
    u32x4 u[2];
};

// round-to-nearest-even fp32 -> bf16 (as raw ushort)
__device__ __forceinline__ unsigned short f2bf(float f) {
    unsigned u = __float_as_uint(f);
    u += 0x7fffu + ((u >> 16) & 1u);
    return (unsigned short)(u >> 16);
}

// hardware tanh if available (gfx1250 V_TANH_F32), else libm
__device__ __forceinline__ float fast_tanh(float x) {
#if __has_builtin(__builtin_amdgcn_tanhf)
    return __builtin_amdgcn_tanhf(x);
#elif __has_builtin(__builtin_amdgcn_tanh_f32)
    return __builtin_amdgcn_tanh_f32(x);
#else
    return tanhf(x);
#endif
}

// Problem constants
#define BATCH 64
#define SEQ   2048
#define DIM   512      // D == O == 512
#define ROWS_TOTAL (BATCH * SEQ)     // 131072
#define MBLK  128                    // rows per block in GEMM kernel
#define SROW  520                    // padded LDS row stride (bf16): 1040B -> 4-dword bank step
#define A_ELEMS (MBLK * SROW)        // 66560 ushorts = 133120 B
#define B_ELEMS 32768                // 64 KB panel: 4 nt * 16 kc * 32 lanes * 16 bf16

// ---------------- async global->LDS staging (CDNA5) ----------------
#if __has_builtin(__builtin_amdgcn_global_load_async_to_lds_b128)
#define HAVE_ASYNC_LDS 1
typedef int v4i __attribute__((ext_vector_type(4)));
typedef const __attribute__((address_space(1))) char* gas1_t;
typedef __attribute__((address_space(3))) char*       las3_t;
typedef __attribute__((address_space(1))) v4i*        gv4i_t;
typedef __attribute__((address_space(3))) v4i*        lv4i_t;
#endif

// Copy one 64 KB B-panel (4096 x 16B) with 256 threads.
__device__ __forceinline__ void stage_panel(const unsigned short* __restrict__ gsrc,
                                            unsigned short* __restrict__ ldst,
                                            int tid) {
#ifdef HAVE_ASYNC_LDS
    gas1_t g = (gas1_t)(const char*)gsrc;
    las3_t l = (las3_t)(char*)ldst;
#pragma unroll 4
    for (int it = 0; it < 16; ++it) {
        int byteOff = (tid + it * 256) << 4;   // *16 bytes
        __builtin_amdgcn_global_load_async_to_lds_b128(
            (gv4i_t)(g + byteOff), (lv4i_t)(l + byteOff), 0, 0);
    }
#else
    const u32x4* src = reinterpret_cast<const u32x4*>(gsrc);
    u32x4* dst = reinterpret_cast<u32x4*>(ldst);
#pragma unroll 4
    for (int it = 0; it < 16; ++it)
        dst[tid + it * 256] = src[tid + it * 256];
#endif
}

__device__ __forceinline__ void wait_async0() {
#ifdef HAVE_ASYNC_LDS
#if __has_builtin(__builtin_amdgcn_s_wait_asynccnt)
    __builtin_amdgcn_s_wait_asynccnt(0);
#else
    asm volatile("s_wait_asynccnt 0x0" ::: "memory");
#endif
#endif
}

// =====================================================================
// Kernel 0: repack W (512x512 fp32, row-major [k][n]) into bf16 WMMA
// B-operand lane layout: block(nt,kc) of 32x16(KxN) -> 32 lanes x 16 bf16
// (32 contiguous bytes per lane).
//   lane l: N = nt*16 + (l&15); base=(l<16?0:8),
//   vgpr v: grp=(v<4?0:16), k_local = base+grp+(v&3)*2 (+1 hi half)
// Linear order [nt][kc][lane][16] -> 4 consecutive nt = contiguous 64KB panel.
// =====================================================================
__global__ __launch_bounds__(256) void k_prep_w(const float* __restrict__ W,
                                                unsigned short* __restrict__ Wpack) {
    int t = blockIdx.x * 256 + threadIdx.x;   // 0 .. 16383
    int lane = t & 31;
    int kc   = (t >> 5) & 15;
    int nt   = t >> 9;
    int n    = (nt << 4) + (lane & 15);
    int kb   = (lane < 16) ? 0 : 8;

    unsigned p[8];
#pragma unroll
    for (int v = 0; v < 8; ++v) {
        int grp = (v < 4) ? 0 : 16;
        int k   = (kc << 5) + kb + grp + ((v & 3) << 1);
        unsigned short lo = f2bf(W[(size_t)k * DIM + n]);
        unsigned short hi = f2bf(W[(size_t)(k + 1) * DIM + n]);
        p[v] = (unsigned)lo | ((unsigned)hi << 16);
    }
    u32x4 d0 = {p[0], p[1], p[2], p[3]};
    u32x4 d1 = {p[4], p[5], p[6], p[7]};
    u32x4* dst = reinterpret_cast<u32x4*>(Wpack + (size_t)t * 16);
    dst[0] = d0;
    dst[1] = d1;
}

// =====================================================================
// Kernel 1: fused  logits[row] = sum_n tanh( (x@W)[row,n] + b[n] ) * wctx[n]
// Block: 256 threads = 8 waves; 128 rows. x tile bf16 in LDS (padded),
// W panels (64 cols) double-buffered in LDS, async-staged.
// =====================================================================
__global__ __launch_bounds__(256) void k_proj_logits(
    const float* __restrict__ x,
    const unsigned short* __restrict__ Wpack,
    const float* __restrict__ bias,
    const float* __restrict__ wctx,
    float* __restrict__ logits)
{
    extern __shared__ unsigned short smem[];
    unsigned short* sA  = smem;                      // [128][SROW] bf16
    unsigned short* sB0 = smem + A_ELEMS;            // panel buffer 0 (64 KB)
    unsigned short* sB1 = sB0 + B_ELEMS;             // panel buffer 1 (64 KB)

    const int tid  = threadIdx.x;
    const int wave = tid >> 5;
    const int lane = tid & 31;
    const size_t rowBase = (size_t)blockIdx.x * MBLK;

    // issue async staging of panel 0 first so it overlaps the A convert below
    stage_panel(Wpack, sB0, tid);

    // ---- stage A: 128 rows x 512 fp32 -> bf16 LDS (coalesced float4 reads) ----
#pragma unroll 4
    for (int it = 0; it < 64; ++it) {
        int u  = tid + it * 256;
        int r  = u >> 7;            // row 0..127
        int c4 = (u & 127) << 2;    // col 0..508 step 4
        const float4* gp = reinterpret_cast<const float4*>(x + (rowBase + r) * DIM + c4);
        float4 v = *gp;
        unsigned p0 = (unsigned)f2bf(v.x) | ((unsigned)f2bf(v.y) << 16);
        unsigned p1 = (unsigned)f2bf(v.z) | ((unsigned)f2bf(v.w) << 16);
        uint2* sp = reinterpret_cast<uint2*>(sA + r * SROW + c4);  // 8B aligned
        *sp = make_uint2(p0, p1);
    }
    wait_async0();
    __syncthreads();   // A staged + panel 0 visible to all waves

    // per-wave A addressing (A-operand 16x32 bf16 layout)
    const int rowLocal = (wave << 4) + (lane & 15);
    const int kbase    = (lane < 16) ? 0 : 8;
    const unsigned short* aBase = sA + rowLocal * SROW + kbase;

    float pl[8] = {0.f, 0.f, 0.f, 0.f, 0.f, 0.f, 0.f, 0.f};

    for (int g = 0; g < 8; ++g) {          // 8 groups of 4 N-tiles (64 cols)
        unsigned short* cur = (g & 1) ? sB1 : sB0;
        unsigned short* nxt = (g & 1) ? sB0 : sB1;

        // prefetch next panel while computing on the current one
        if (g < 7) stage_panel(Wpack + (size_t)(g + 1) * B_ELEMS, nxt, tid);

        v8f acc[4] = {v8f{}, v8f{}, v8f{}, v8f{}};

        for (int kc = 0; kc < 16; ++kc) {
            BF16x16 a;
            const unsigned short* ap = aBase + (kc << 5);
            a.u[0] = *reinterpret_cast<const u32x4*>(ap);        // K 0..7  (or 8..15)
            a.u[1] = *reinterpret_cast<const u32x4*>(ap + 16);   // K 16..23 (or 24..31)
#pragma unroll
            for (int j = 0; j < 4; ++j) {
                BF16x16 b;
                const unsigned short* bp = cur + ((((j << 4) + kc) << 5) + lane) * 16;
                b.u[0] = *reinterpret_cast<const u32x4*>(bp);
                b.u[1] = *reinterpret_cast<const u32x4*>(bp + 8);
                acc[j] = __builtin_amdgcn_wmma_f32_16x16x32_bf16(
                    false, a.bf, false, b.bf, (short)0, acc[j], false, false);
            }
        }

        // ---- epilogue: tanh(+bias) * wctx, accumulate per-row partials ----
        // C layout: lane&15 = N within tile; vgpr r = row r (lanes<16) / r+8 (lanes>=16)
#pragma unroll
        for (int j = 0; j < 4; ++j) {
            int n = (((g << 2) + j) << 4) + (lane & 15);
            float bv = bias[n];
            float wv = wctx[n];
#pragma unroll
            for (int r = 0; r < 8; ++r)
                pl[r] += fast_tanh(acc[j][r] + bv) * wv;
        }

        // next panel staged (my share) + all waves done reading `nxt`'s old data
        wait_async0();
        __syncthreads();
    }

    // reduce over the 16 lanes of each half (N dimension)
#pragma unroll
    for (int r = 0; r < 8; ++r) {
        float v = pl[r];
        v += __shfl_xor(v, 1, 32);
        v += __shfl_xor(v, 2, 32);
        v += __shfl_xor(v, 4, 32);
        v += __shfl_xor(v, 8, 32);
        pl[r] = v;
    }
    if (lane == 0 || lane == 16) {
        size_t row = rowBase + (size_t)(wave << 4) + ((lane >> 4) << 3);
#pragma unroll
        for (int r = 0; r < 8; ++r)
            logits[row + r] = pl[r];
    }
}

// =====================================================================
// Kernel 2: per-batch softmax over S=2048 (in place) + zero out[b,:]
// =====================================================================
__global__ __launch_bounds__(256) void k_softmax_zero(float* __restrict__ logits,
                                                      float* __restrict__ out) {
    __shared__ float red[256];
    const int b = blockIdx.x, tid = threadIdx.x;
    float v[8];
    float m = -3.4e38f;
#pragma unroll
    for (int i = 0; i < 8; ++i) {
        v[i] = logits[b * SEQ + tid + i * 256];
        m = fmaxf(m, v[i]);
    }
    red[tid] = m; __syncthreads();
    for (int s = 128; s > 0; s >>= 1) {
        if (tid < s) red[tid] = fmaxf(red[tid], red[tid + s]);
        __syncthreads();
    }
    m = red[0]; __syncthreads();

    float sum = 0.f;
#pragma unroll
    for (int i = 0; i < 8; ++i) { v[i] = __expf(v[i] - m); sum += v[i]; }
    red[tid] = sum; __syncthreads();
    for (int s = 128; s > 0; s >>= 1) {
        if (tid < s) red[tid] += red[tid + s];
        __syncthreads();
    }
    float inv = 1.0f / red[0];
#pragma unroll
    for (int i = 0; i < 8; ++i)
        logits[b * SEQ + tid + i * 256] = v[i] * inv;

    // zero output rows (required each replay: kernel 3 uses atomics)
    out[b * DIM + tid] = 0.f;
    out[b * DIM + tid + 256] = 0.f;
}

// =====================================================================
// Kernel 3: out[b,d] += sum_s attn[b,s] * x[b,s,d]   (streaming, 268 MB)
// grid (16 s-chunks, 64 batches), 512 threads = all d columns
// =====================================================================
__global__ __launch_bounds__(512) void k_wsum(const float* __restrict__ x,
                                              const float* __restrict__ attn,
                                              float* __restrict__ out) {
    __shared__ float sa[128];
    const int b = blockIdx.y, sc = blockIdx.x, tid = threadIdx.x;
    if (tid < 128) sa[tid] = attn[b * SEQ + sc * 128 + tid];
    __syncthreads();
    float acc = 0.f;
    const float* xp = x + ((size_t)b * SEQ + sc * 128) * DIM + tid;
#pragma unroll 4
    for (int i = 0; i < 128; ++i)
        acc += sa[i] * xp[(size_t)i * DIM];
    atomicAdd(&out[b * DIM + tid], acc);
}

// =====================================================================
extern "C" void kernel_launch(void* const* d_in, const int* in_sizes, int n_in,
                              void* d_out, int out_size, void* d_ws, size_t ws_size,
                              hipStream_t stream) {
    const float* x      = (const float*)d_in[0];   // [64,2048,512]
    const float* W_proj = (const float*)d_in[1];   // [512,512]
    const float* b_proj = (const float*)d_in[2];   // [512]
    const float* w_ctx  = (const float*)d_in[3];   // [512]
    float* out = (float*)d_out;                    // [64,512]

    // workspace layout
    unsigned short* Wpack = (unsigned short*)d_ws;              // 512 KB
    float* logits = (float*)((char*)d_ws + 512 * 1024);         // 512 KB (attn in-place)

    // 0) repack W into bf16 WMMA lane layout
    k_prep_w<<<64, 256, 0, stream>>>(W_proj, Wpack);

    // 1) fused GEMM + tanh + context dot -> logits (bf16 WMMA, async staging)
    const size_t smemBytes = (size_t)(A_ELEMS + 2 * B_ELEMS) * sizeof(unsigned short); // 264192 B
    k_proj_logits<<<ROWS_TOTAL / MBLK, 256, smemBytes, stream>>>(
        x, Wpack, b_proj, w_ctx, logits);

    // 2) softmax over S per batch (in place) + zero d_out
    k_softmax_zero<<<BATCH, 256, 0, stream>>>(logits, out);

    // 3) weighted sum of original x
    k_wsum<<<dim3(16, BATCH), 512, 0, stream>>>(x, logits, out);
}
